// TransformerLayer_283467842150
// MI455X (gfx1250) — matrix operand order repaired and checked by reading, not compiled, not run on hardware
//
#include <hip/hip_runtime.h>
#include <hip/hip_bf16.h>
#include <math.h>

typedef __attribute__((ext_vector_type(16))) _Float16 v16h;
typedef __attribute__((ext_vector_type(8)))  _Float16 v8h;
typedef __attribute__((ext_vector_type(8)))  float    v8f;

#define D_MODEL 256
#define B_SZ    4
#define HW      128
#define L_TOK   (HW*HW)        // 16384 tokens per batch image
#define NTOK    (B_SZ*L_TOK)   // 65536 total tokens
#define WT      1024           // tokens per window (32x32)
#define NWIN    64             // b * ns * ns
#define FFN_IN  512
#define FFN_HID 2048
#define LN_EPS  1e-5f

// ---------------------------------------------------------------------------
// Window mapping: window w = (b, i, j); token t = (r, s) inside 32x32 window.
// Rolled by (-16,-16) then partitioned -> original l index.
// ---------------------------------------------------------------------------
__device__ __forceinline__ int window_row_to_l(int w, int t) {
    int b = w >> 4, widx = w & 15;
    int i = widx >> 2, j = widx & 3;
    int r = t >> 5, s = t & 31;
    int oh = (i * 32 + r + 16) & 127;
    int ow = (j * 32 + s + 16) & 127;
    return b * L_TOK + oh * HW + ow;
}

// ---------------------------------------------------------------------------
// WMMA fragment loaders (contiguous k = 16h + s order for A and B, f16 16x16x32)
// A (16xK): lane<16 -> row=lane, K kb..kb+15 contiguous
//           lane>=16 -> row=lane-16, K kb+16..kb+31 contiguous
// ---------------------------------------------------------------------------
__device__ __forceinline__ v16h load_a_frag(const _Float16* row, int kbase, int lane) {
    int off = kbase + ((lane >> 4) << 4);
    v8h lo = *(const v8h*)(row + off);
    v8h hi = *(const v8h*)(row + off + 8);
    v16h a;
#pragma unroll
    for (int i = 0; i < 8; ++i) { a[i] = lo[i]; a[i + 8] = hi[i]; }
    return a;
}

// B (Kx16): lane = column (lane&15); lane<16 holds K kb..kb+15 contiguous,
// lane>=16 holds K kb+16..kb+31 (mirrors sparse-B doc layout).
__device__ __forceinline__ v16h load_b_frag(const _Float16* mat, int ldk,
                                            int ncol_base, int kbase, int lane) {
    int ncol = ncol_base + (lane & 15);
    const _Float16* p = mat + (size_t)ncol * ldk + kbase + ((lane >> 4) << 4);
    return *(const v16h*)p;
}

__device__ __forceinline__ v8f wmma_f16(v16h a, v16h b, v8f c) {
    return __builtin_amdgcn_wmma_f32_16x16x32_f16(false, a, false, b,
                                                  (short)0, c, false, false);
}

// ---------------------------------------------------------------------------
// f32 -> f16 conversion
// ---------------------------------------------------------------------------
__global__ void cvt_kernel(const float* __restrict__ in, _Float16* __restrict__ out,
                           int nElem) {
    int i = blockIdx.x * blockDim.x + threadIdx.x;
    if (i < nElem) out[i] = (_Float16)in[i];
}

// ---------------------------------------------------------------------------
// QKV projection, gathered directly into windowed order.
// q_w,k_w: [win][1024][256] f16; v stored transposed vt: [win][256][1024] f16
// block = 8 waves; block handles 16 tokens x 256 channels for q, k, v.
// Target A-fragments are loaded once and reused for both K and V GEMMs.
// ---------------------------------------------------------------------------
__global__ __launch_bounds__(256) void qkv_kernel(
    const _Float16* __restrict__ srch, const _Float16* __restrict__ tgth,
    const _Float16* __restrict__ Wqh, const _Float16* __restrict__ Wkh,
    const _Float16* __restrict__ Wvh,
    _Float16* __restrict__ qw, _Float16* __restrict__ kw, _Float16* __restrict__ vt)
{
    int blk = blockIdx.x;
    int w = blk >> 6, mt = blk & 63;
    int t0 = mt << 4;
    int lane = threadIdx.x & 31, wave = threadIdx.x >> 5;
    int hi = lane >> 4, n = lane & 15;
    int l = window_row_to_l(w, t0 + n);      // gathered A row for this lane
    int n0 = wave * 32;

    const _Float16* srow = srch + (size_t)l * D_MODEL;
    const _Float16* trow = tgth + (size_t)l * D_MODEL;

    // ---- Q = src @ Wq^T ----
    {
        v8f acc0 = {}, acc1 = {};
#pragma unroll
        for (int ks = 0; ks < 8; ++ks) {
            int kb = ks * 32;
            v16h a  = load_a_frag(srow, kb, lane);
            v16h b0 = load_b_frag(Wqh, D_MODEL, n0,      kb, lane);
            v16h b1 = load_b_frag(Wqh, D_MODEL, n0 + 16, kb, lane);
            acc0 = wmma_f16(a, b0, acc0);
            acc1 = wmma_f16(a, b1, acc1);
        }
#pragma unroll
        for (int r = 0; r < 8; ++r) {
            int row = t0 + r + hi * 8;
            qw[((size_t)w * WT + row) * D_MODEL + n0 + n]      = (_Float16)acc0[r];
            qw[((size_t)w * WT + row) * D_MODEL + n0 + 16 + n] = (_Float16)acc1[r];
        }
    }

    // ---- target A fragments: load once, use for K and V ----
    v16h tf[8];
#pragma unroll
    for (int ks = 0; ks < 8; ++ks) tf[ks] = load_a_frag(trow, ks * 32, lane);

    // ---- K = tgt @ Wk^T ----
    {
        v8f acc0 = {}, acc1 = {};
#pragma unroll
        for (int ks = 0; ks < 8; ++ks) {
            int kb = ks * 32;
            v16h b0 = load_b_frag(Wkh, D_MODEL, n0,      kb, lane);
            v16h b1 = load_b_frag(Wkh, D_MODEL, n0 + 16, kb, lane);
            acc0 = wmma_f16(tf[ks], b0, acc0);
            acc1 = wmma_f16(tf[ks], b1, acc1);
        }
#pragma unroll
        for (int r = 0; r < 8; ++r) {
            int row = t0 + r + hi * 8;
            kw[((size_t)w * WT + row) * D_MODEL + n0 + n]      = (_Float16)acc0[r];
            kw[((size_t)w * WT + row) * D_MODEL + n0 + 16 + n] = (_Float16)acc1[r];
        }
    }

    // ---- V = tgt @ Wv^T (stored transposed) ----
    {
        v8f acc0 = {}, acc1 = {};
#pragma unroll
        for (int ks = 0; ks < 8; ++ks) {
            int kb = ks * 32;
            v16h b0 = load_b_frag(Wvh, D_MODEL, n0,      kb, lane);
            v16h b1 = load_b_frag(Wvh, D_MODEL, n0 + 16, kb, lane);
            acc0 = wmma_f16(tf[ks], b0, acc0);
            acc1 = wmma_f16(tf[ks], b1, acc1);
        }
#pragma unroll
        for (int r = 0; r < 8; ++r) {
            int row = t0 + r + hi * 8;
            vt[((size_t)w * D_MODEL + n0 + n) * WT + row]      = (_Float16)acc0[r];
            vt[((size_t)w * D_MODEL + n0 + 16 + n) * WT + row] = (_Float16)acc1[r];
        }
    }
}

// ---------------------------------------------------------------------------
// Flash-style windowed attention. Block = 16 queries of one window, 8 waves.
// Q fragments are register-resident across the whole 1024-key sweep.
// ---------------------------------------------------------------------------
__global__ __launch_bounds__(256) void attn_kernel(
    const _Float16* __restrict__ qw, const _Float16* __restrict__ kw,
    const _Float16* __restrict__ vt, const float* __restrict__ mask,
    _Float16* __restrict__ attnout)
{
    __shared__ float    S[16][128];
    __shared__ _Float16 P[16][128];
    __shared__ float    rowm[16], rowl[16], rowc[16];

    int blk = blockIdx.x;
    int w = blk >> 6, mt = blk & 63;
    int t0 = mt << 4;
    int widx = w & 15;
    int lane = threadIdx.x & 31, wave = threadIdx.x >> 5;
    int hi = lane >> 4, n = lane & 15;
    int ch0 = wave * 32;

    if (threadIdx.x < 16) { rowm[threadIdx.x] = -1e30f; rowl[threadIdx.x] = 0.f; }
    __syncthreads();

    const _Float16* qrow   = qw + ((size_t)w * WT + t0 + n) * D_MODEL;
    const _Float16* kbase  = kw + (size_t)w * WT * D_MODEL;
    const _Float16* vbase0 = vt + (size_t)w * D_MODEL * WT;

    // Q A-fragments: loaded once, reused for all 8 key chunks (64 VGPRs)
    v16h qf[8];
#pragma unroll
    for (int ks = 0; ks < 8; ++ks) qf[ks] = load_a_frag(qrow, ks * 32, lane);

    v8f accO0 = {}, accO1 = {};

    for (int c = 0; c < 8; ++c) {
        int kt0 = c * 128;
        int keycol = kt0 + wave * 16;
        // S tile: 16 queries x 16 keys, K=256
        v8f s = {};
#pragma unroll
        for (int ks = 0; ks < 8; ++ks) {
            v16h b = load_b_frag(kbase, D_MODEL, keycol, ks * 32, lane);
            s = wmma_f16(qf[ks], b, s);
        }
        // scale + mask -> LDS
#pragma unroll
        for (int r = 0; r < 8; ++r) {
            int qr = r + hi * 8;
            int key = keycol + n;
            float val = s[r] * 0.0625f +
                        mask[((size_t)widx * WT + (t0 + qr)) * WT + key];
            S[qr][wave * 16 + n] = val;
        }
        __syncthreads();
        // online softmax update (one thread per query row)
        if (threadIdx.x < 16) {
            int q = threadIdx.x;
            float mo = rowm[q];
            float cm = -1e30f;
            for (int kk = 0; kk < 128; ++kk) cm = fmaxf(cm, S[q][kk]);
            float mn = fmaxf(mo, cm);
            float corr = __expf(mo - mn);
            float sum = 0.f;
            for (int kk = 0; kk < 128; ++kk) {
                float p = __expf(S[q][kk] - mn);
                P[q][kk] = (_Float16)p;
                sum += p;
            }
            rowm[q] = mn;
            rowl[q] = rowl[q] * corr + sum;
            rowc[q] = corr;
        }
        __syncthreads();
        // rescale running O, then O += P @ V(chunk)
#pragma unroll
        for (int r = 0; r < 8; ++r) {
            float corr = rowc[r + hi * 8];
            accO0[r] *= corr; accO1[r] *= corr;
        }
        const _Float16* prow  = &P[n][0];
        const _Float16* vbase = vbase0 + kt0;
#pragma unroll
        for (int ks = 0; ks < 4; ++ks) {
            v16h a  = load_a_frag(prow, ks * 32, lane);
            v16h b0 = load_b_frag(vbase, WT, ch0,      ks * 32, lane);
            v16h b1 = load_b_frag(vbase, WT, ch0 + 16, ks * 32, lane);
            accO0 = wmma_f16(a, b0, accO0);
            accO1 = wmma_f16(a, b1, accO1);
        }
        __syncthreads();  // P consumed before next chunk rewrites it
    }
    // normalize and scatter to l-order
#pragma unroll
    for (int r = 0; r < 8; ++r) {
        int qr = r + hi * 8;
        float inv = 1.0f / rowl[qr];
        int l = window_row_to_l(w, t0 + qr);
        attnout[(size_t)l * D_MODEL + ch0 + n]      = (_Float16)(accO0[r] * inv);
        attnout[(size_t)l * D_MODEL + ch0 + 16 + n] = (_Float16)(accO1[r] * inv);
    }
}

// ---------------------------------------------------------------------------
// message = LN(attn_out @ Wm^T) -> f16 (feeds FFN concat channels 256..511)
// ---------------------------------------------------------------------------
__global__ __launch_bounds__(256) void msg_kernel(
    const _Float16* __restrict__ attnout, const _Float16* __restrict__ Wmh,
    const float* __restrict__ g1, const float* __restrict__ b1,
    _Float16* __restrict__ msgh)
{
    __shared__ float buf[16][D_MODEL];
    __shared__ float mean_s[16], istd_s[16];
    int l0 = blockIdx.x << 4;
    int lane = threadIdx.x & 31, wave = threadIdx.x >> 5;
    int hi = lane >> 4, n = lane & 15;
    int n0 = wave * 32;

    const _Float16* arow = attnout + (size_t)(l0 + n) * D_MODEL;
    v8f acc0 = {}, acc1 = {};
#pragma unroll
    for (int ks = 0; ks < 8; ++ks) {
        int kb = ks * 32;
        v16h a  = load_a_frag(arow, kb, lane);
        v16h b0 = load_b_frag(Wmh, D_MODEL, n0,      kb, lane);
        v16h b1 = load_b_frag(Wmh, D_MODEL, n0 + 16, kb, lane);
        acc0 = wmma_f16(a, b0, acc0);
        acc1 = wmma_f16(a, b1, acc1);
    }
#pragma unroll
    for (int r = 0; r < 8; ++r) {
        buf[r + hi * 8][n0 + n]      = acc0[r];
        buf[r + hi * 8][n0 + 16 + n] = acc1[r];
    }
    __syncthreads();
    if (threadIdx.x < 16) {
        int q = threadIdx.x;
        float m = 0.f;
        for (int c = 0; c < D_MODEL; ++c) m += buf[q][c];
        m *= (1.0f / D_MODEL);
        float v = 0.f;
        for (int c = 0; c < D_MODEL; ++c) { float d = buf[q][c] - m; v += d * d; }
        v *= (1.0f / D_MODEL);
        mean_s[q] = m;
        istd_s[q] = rsqrtf(v + LN_EPS);
    }
    __syncthreads();
    int row = threadIdx.x >> 4, c0 = (threadIdx.x & 15) << 4;
    float m = mean_s[row], is = istd_s[row];
#pragma unroll
    for (int cc = 0; cc < 16; ++cc) {
        int c = c0 + cc;
        float v = (buf[row][c] - m) * is * g1[c] + b1[c];
        msgh[(size_t)(l0 + row) * D_MODEL + c] = (_Float16)v;
    }
}

// ---------------------------------------------------------------------------
// Fused FFN: hdn = gelu([src|msg] @ W1^T) (kept in LDS, two 1024-wide halves),
// y = LN(hdn @ W2^T) ; out = source + y  (f32)
// ---------------------------------------------------------------------------
__global__ __launch_bounds__(256) void ffn_kernel(
    const _Float16* __restrict__ srch, const _Float16* __restrict__ msgh,
    const _Float16* __restrict__ W1h, const _Float16* __restrict__ W2h,
    const float* __restrict__ g2, const float* __restrict__ b2,
    const float* __restrict__ source, float* __restrict__ out)
{
    __shared__ _Float16 hdn[16][1024];   // 32 KB: half of the hidden activation
    __shared__ float    buf[16][D_MODEL];
    __shared__ float    mean_s[16], istd_s[16];

    int l0 = blockIdx.x << 4;
    int lane = threadIdx.x & 31, wave = threadIdx.x >> 5;
    int hi = lane >> 4, n = lane & 15;
    int co0 = wave * 32;

    const _Float16* srow = srch + (size_t)(l0 + n) * D_MODEL;
    const _Float16* mrow = msgh + (size_t)(l0 + n) * D_MODEL;
    v8f acc0 = {}, acc1 = {};

    for (int half = 0; half < 2; ++half) {
        // GEMM1: this wave produces 128 hidden cols (4 chunks of 32), K = 512
        for (int chunk = 0; chunk < 4; ++chunk) {
            int h0 = half * 1024 + wave * 128 + chunk * 32;
            // prefetch next chunk's W1 rows into cache (global_prefetch_b8)
            __builtin_prefetch(W1h + (size_t)(h0 + 32 + n) * FFN_IN, 0, 3);
            v8f h0a = {}, h1a = {};
#pragma unroll
            for (int ks = 0; ks < 16; ++ks) {
                int kb = ks * 32;
                const _Float16* arow = (kb < D_MODEL) ? srow : mrow;
                int kloc = (kb < D_MODEL) ? kb : kb - D_MODEL;
                v16h a  = load_a_frag(arow, kloc, lane);
                v16h b0 = load_b_frag(W1h, FFN_IN, h0,      kb, lane);
                v16h b1 = load_b_frag(W1h, FFN_IN, h0 + 16, kb, lane);
                h0a = wmma_f16(a, b0, h0a);
                h1a = wmma_f16(a, b1, h1a);
            }
            int lc = wave * 128 + chunk * 32;
#pragma unroll
            for (int r = 0; r < 8; ++r) {
                int row = r + hi * 8;
                float x0 = h0a[r], x1 = h1a[r];
                x0 = 0.5f * x0 * (1.f + erff(x0 * 0.70710678f));   // exact gelu
                x1 = 0.5f * x1 * (1.f + erff(x1 * 0.70710678f));
                hdn[row][lc + n]      = (_Float16)x0;
                hdn[row][lc + 16 + n] = (_Float16)x1;
            }
        }
        __syncthreads();
        // GEMM2 partial: accumulate K = this half's 1024 hidden dims
        const _Float16* prow = &hdn[n][0];
        // prefetch the other half's W2 columns while we consume this one
        __builtin_prefetch(W2h + (size_t)(co0 + n) * FFN_HID + (1 - half) * 1024, 0, 3);
#pragma unroll 8
        for (int ks = 0; ks < 32; ++ks) {
            int kb = ks * 32;
            v16h a  = load_a_frag(prow, kb, lane);
            v16h b0 = load_b_frag(W2h, FFN_HID, co0,      half * 1024 + kb, lane);
            v16h b1 = load_b_frag(W2h, FFN_HID, co0 + 16, half * 1024 + kb, lane);
            acc0 = wmma_f16(a, b0, acc0);
            acc1 = wmma_f16(a, b1, acc1);
        }
        __syncthreads();
    }
    // LN + residual
#pragma unroll
    for (int r = 0; r < 8; ++r) {
        buf[r + hi * 8][co0 + n]      = acc0[r];
        buf[r + hi * 8][co0 + 16 + n] = acc1[r];
    }
    __syncthreads();
    if (threadIdx.x < 16) {
        int q = threadIdx.x;
        float m = 0.f;
        for (int c = 0; c < D_MODEL; ++c) m += buf[q][c];
        m *= (1.0f / D_MODEL);
        float v = 0.f;
        for (int c = 0; c < D_MODEL; ++c) { float d = buf[q][c] - m; v += d * d; }
        v *= (1.0f / D_MODEL);
        mean_s[q] = m;
        istd_s[q] = rsqrtf(v + LN_EPS);
    }
    __syncthreads();
    int row = threadIdx.x >> 4, c0 = (threadIdx.x & 15) << 4;
    float m = mean_s[row], is = istd_s[row];
#pragma unroll
    for (int cc = 0; cc < 16; ++cc) {
        int c = c0 + cc;
        float v = (buf[row][c] - m) * is * g2[c] + b2[c];
        size_t idx = (size_t)(l0 + row) * D_MODEL + c;
        out[idx] = source[idx] + v;
    }
}

// ---------------------------------------------------------------------------
extern "C" void kernel_launch(void* const* d_in, const int* in_sizes, int n_in,
                              void* d_out, int out_size, void* d_ws, size_t ws_size,
                              hipStream_t stream) {
    (void)in_sizes; (void)n_in; (void)out_size; (void)ws_size;
    const float* source = (const float*)d_in[0];
    const float* target = (const float*)d_in[1];
    const float* Wq = (const float*)d_in[2];
    const float* Wk = (const float*)d_in[3];
    const float* Wv = (const float*)d_in[4];
    const float* Wm = (const float*)d_in[5];
    const float* g1 = (const float*)d_in[6];
    const float* b1 = (const float*)d_in[7];
    const float* W1 = (const float*)d_in[8];
    const float* W2 = (const float*)d_in[9];
    const float* g2 = (const float*)d_in[10];
    const float* b2 = (const float*)d_in[11];
    const float* mask = (const float*)d_in[12];
    float* out = (float*)d_out;

    char* ws = (char*)d_ws;
    size_t off = 0;
    auto alloc_h = [&](size_t halfs) {
        _Float16* p = (_Float16*)(ws + off);
        off += halfs * sizeof(_Float16);
        return p;
    };
    _Float16* srch = alloc_h((size_t)NTOK * D_MODEL);
    _Float16* tgth = alloc_h((size_t)NTOK * D_MODEL);
    _Float16* Wqh  = alloc_h((size_t)D_MODEL * D_MODEL);
    _Float16* Wkh  = alloc_h((size_t)D_MODEL * D_MODEL);
    _Float16* Wvh  = alloc_h((size_t)D_MODEL * D_MODEL);
    _Float16* Wmh  = alloc_h((size_t)D_MODEL * D_MODEL);
    _Float16* W1h  = alloc_h((size_t)FFN_HID * FFN_IN);
    _Float16* W2h  = alloc_h((size_t)D_MODEL * FFN_HID);
    _Float16* qwb  = alloc_h((size_t)NWIN * WT * D_MODEL);
    _Float16* kwb  = alloc_h((size_t)NWIN * WT * D_MODEL);
    _Float16* vtb  = alloc_h((size_t)NWIN * D_MODEL * WT);
    _Float16* aout = alloc_h((size_t)NTOK * D_MODEL);
    _Float16* msgh = alloc_h((size_t)NTOK * D_MODEL);

    auto cvt = [&](const float* in, _Float16* dst, int nElem) {
        cvt_kernel<<<(nElem + 255) / 256, 256, 0, stream>>>(in, dst, nElem);
    };
    cvt(source, srch, NTOK * D_MODEL);
    cvt(target, tgth, NTOK * D_MODEL);
    cvt(Wq, Wqh, D_MODEL * D_MODEL);
    cvt(Wk, Wkh, D_MODEL * D_MODEL);
    cvt(Wv, Wvh, D_MODEL * D_MODEL);
    cvt(Wm, Wmh, D_MODEL * D_MODEL);
    cvt(W1, W1h, FFN_HID * FFN_IN);
    cvt(W2, W2h, D_MODEL * FFN_HID);

    qkv_kernel<<<NWIN * (WT / 16), 256, 0, stream>>>(srch, tgth, Wqh, Wkh, Wvh,
                                                     qwb, kwb, vtb);
    attn_kernel<<<NWIN * (WT / 16), 256, 0, stream>>>(qwb, kwb, vtb, mask, aout);
    msg_kernel<<<NTOK / 16, 256, 0, stream>>>(aout, Wmh, g1, b1, msgh);
    ffn_kernel<<<NTOK / 16, 256, 0, stream>>>(srch, msgh, W1h, W2h, g2, b2,
                                              source, out);
}
